// SchrodingerPinn_21406117003343
// MI455X (gfx1250) — matrix operand order repaired and checked by reading, not compiled, not run on hardware
//
#include <hip/hip_runtime.h>

typedef __attribute__((ext_vector_type(16))) _Float16 v16h;
typedef __attribute__((ext_vector_type(8)))  float    v8f;

union Frag16 { v16h h; uint4 u[2]; };

#define NW   8    // waves per block (256 threads -> 2 waves per SIMD32)
#define WAVE 32   // gfx1250 is wave32

// Branch-free tanh. Prefer the native CDNA5 V_TANH_F32 transcendental; fall
// back to an exp2-based rational form with clamping (no divergent branches).
__device__ __forceinline__ float fast_tanh(float x) {
#if __has_builtin(__builtin_amdgcn_tanhf)
  return __builtin_amdgcn_tanhf(x);
#elif __has_builtin(__builtin_amdgcn_tanh_f32)
  return __builtin_amdgcn_tanh_f32(x);
#else
  float xc = __builtin_fminf(__builtin_fmaxf(x, -10.0f), 10.0f);
  float e  = __builtin_amdgcn_exp2f(xc * 2.8853900817779268f);  // exp(2x)
  return (e - 1.0f) * __builtin_amdgcn_rcpf(e + 1.0f);
#endif
}

// ---------------------------------------------------------------------------
// Prep: convert hidden weights (transposed, [out][in]) and padded w5 to f16.
// wpack layout (halfs): [4][128][128] WT for w1..w4, then [16][128] WT5 (pad 0)
// ---------------------------------------------------------------------------
__global__ void prep_weights_kernel(const float* __restrict__ w1,
                                    const float* __restrict__ w2,
                                    const float* __restrict__ w3,
                                    const float* __restrict__ w4,
                                    const float* __restrict__ w5,
                                    _Float16* __restrict__ wpack) {
  int i = blockIdx.x * blockDim.x + threadIdx.x;
  if (i < 4 * 128 * 128) {
    int l = i >> 14, r = i & 16383, o = r >> 7, in = r & 127;
    const float* w = (l == 0) ? w1 : (l == 1) ? w2 : (l == 2) ? w3 : w4;
    wpack[i] = (_Float16)w[in * 128 + o];          // transpose: [in][out] -> [out][in]
  } else if (i < 4 * 128 * 128 + 16 * 128) {
    int r = i - 4 * 128 * 128, o = r >> 7, in = r & 127;
    wpack[i] = (_Float16)((o < 2) ? w5[in * 2 + o] : 0.0f);   // 128x2 padded to 128x16
  }
}

// ---------------------------------------------------------------------------
// Main PINN MLP kernel. S = number of forward-mode streams:
//   stream 0 = value, 1 = d/dx, 2 = d/dt, 3 = d2/dx2
// S==4: equation points (writes h and residual)
// S==2: boundary points (writes h and dh/dx)
// S==1: init points    (writes h)
// Each wave handles one 16-point tile. WMMA: D(16x16) = A(16pts x 32K f16) x
// B(32K x 16 outF f16) + C(f32), accumulated over 4 K-chunks, 8 N-tiles.
// ---------------------------------------------------------------------------
template <int S>
__global__ __launch_bounds__(NW * WAVE) void pinn_mlp_kernel(
    const float* __restrict__ tx,
    const _Float16* __restrict__ wpack,
    const float* __restrict__ w0, const float* __restrict__ b0,
    const float* __restrict__ b1, const float* __restrict__ b2,
    const float* __restrict__ b3, const float* __restrict__ b4,
    const float* __restrict__ b5,
    float* __restrict__ out_h, float* __restrict__ out_aux,
    int ntiles) {
  extern __shared__ char smem[];
  _Float16* wlds   = (_Float16*)smem;            // [4][128][128]  f16 WT (hidden)
  _Float16* w5lds  = wlds + 4 * 128 * 128;       // [16][128]      f16 WT5 padded
  float*    fbase  = (float*)(w5lds + 16 * 128);
  float*    biaslds = fbase;                     // [4][128]  b1..b4
  float*    b5lds  = fbase + 512;                // [2] (pad 4)
  float*    w0lds  = fbase + 516;                // [2][128]
  float*    b0lds  = fbase + 772;                // [128]
  _Float16* actall = (_Float16*)(fbase + 900);   // [NW][S][16][128] f16 activations
  float*    stageall = (float*)(actall + NW * S * 16 * 128); // [NW][S][16][2]

  const int tid = threadIdx.x;

  // Cooperative copy of packed f16 weights: (65536+2048)*2B / 16B = 8448 uint4
  {
    const uint4* src = (const uint4*)wpack;
    uint4* dst = (uint4*)wlds;
    for (int i = tid; i < 8448; i += NW * WAVE) dst[i] = src[i];
    for (int i = tid; i < 128; i += NW * WAVE) {
      biaslds[i]       = b1[i];
      biaslds[128 + i] = b2[i];
      biaslds[256 + i] = b3[i];
      biaslds[384 + i] = b4[i];
      b0lds[i] = b0[i];
      w0lds[i] = w0[i];            // w0 row 0: dz/dt weights
      w0lds[128 + i] = w0[128 + i];// w0 row 1: dz/dx weights
    }
    if (tid < 2) b5lds[tid] = b5[tid];
  }
  __syncthreads();

  const int wave = tid >> 5;
  const int lane = tid & 31;
  const int m  = lane & 15;   // row / column selector within 16-lane half
  const int lh = lane >> 4;   // lane-half
  _Float16* act  = actall + wave * (S * 16 * 128);
  float*    stage = stageall + wave * (S * 16 * 2);

  for (int tile = blockIdx.x * NW + wave; tile < ntiles; tile += gridDim.x * NW) {
    const int pbase = tile * 16;

    // speculative prefetch of the next grid-stride tile's inputs
    if (lane == 0) {
      const int nbase = pbase + (int)gridDim.x * NW * 16;
      __builtin_prefetch(tx + nbase * 2, 0, 1);
    }

    // ---------------- input layer (2 -> 128), VALU ----------------
    // p is wave-uniform -> tx loads scalarize to SMEM; features lane-striped.
    for (int p = 0; p < 16; ++p) {
      const float t = tx[(pbase + p) * 2 + 0];
      const float x = tx[(pbase + p) * 2 + 1];
      #pragma unroll
      for (int k = 0; k < 4; ++k) {
        const int f = k * 32 + lane;
        float wt0 = w0lds[f], wx0 = w0lds[128 + f];
        float z = fmaf(t, wt0, fmaf(x, wx0, b0lds[f]));
        float a = fast_tanh(z);
        float s1 = 1.0f - a * a;
        act[(0 * 16 + p) * 128 + f] = (_Float16)a;
        if constexpr (S >= 2) {
          float ax = s1 * wx0;                         // z_x = wx0, z_xx = 0
          act[(1 * 16 + p) * 128 + f] = (_Float16)ax;
          if constexpr (S == 4) {
            act[(2 * 16 + p) * 128 + f] = (_Float16)(s1 * wt0);
            act[(3 * 16 + p) * 128 + f] = (_Float16)(-2.0f * a * wx0 * ax);
          }
        }
      }
    }

    // ---------------- hidden layers (128 -> 128) x4, WMMA ----------------
    for (int l = 0; l < 4; ++l) {
      Frag16 afr[S][4];  // A operands: [stream][K-chunk], loaded before overwrite
      #pragma unroll
      for (int s = 0; s < S; ++s) {
        #pragma unroll
        for (int c = 0; c < 4; ++c) {
          const _Float16* ap = act + (s * 16 + m) * 128 + c * 32 + 8 * lh;
          afr[s][c].u[0] = *(const uint4*)ap;         // K = 8*lh + 0..7
          afr[s][c].u[1] = *(const uint4*)(ap + 16);  // K = 16 + 8*lh + 0..7
        }
      }
      const _Float16* wrow = wlds + l * 16384;
      for (int nt = 0; nt < 8; ++nt) {
        const int col = nt * 16 + m;  // output feature of this lane's D column
        v8f accv, acc1{}, acc2{}, acc3{};
        float bv = biaslds[l * 128 + col];
        #pragma unroll
        for (int r = 0; r < 8; ++r) accv[r] = bv;     // bias only in value stream
        const _Float16* wcol = wrow + col * 128 + 8 * lh;
        #pragma unroll
        for (int c = 0; c < 4; ++c) {
          Frag16 bf;  // B operand: K = c*32 + 8*lh + 0..7, then c*32 + 16 + 8*lh + 0..7
          const _Float16* bp = wcol + c * 32;
          bf.u[0] = *(const uint4*)bp;
          bf.u[1] = *(const uint4*)(bp + 16);
          accv = __builtin_amdgcn_wmma_f32_16x16x32_f16(false, afr[0][c].h, false, bf.h,
                                                        (short)0, accv, false, false);
          if constexpr (S >= 2)
            acc1 = __builtin_amdgcn_wmma_f32_16x16x32_f16(false, afr[1][c].h, false, bf.h,
                                                          (short)0, acc1, false, false);
          if constexpr (S == 4) {
            acc2 = __builtin_amdgcn_wmma_f32_16x16x32_f16(false, afr[2][c].h, false, bf.h,
                                                          (short)0, acc2, false, false);
            acc3 = __builtin_amdgcn_wmma_f32_16x16x32_f16(false, afr[3][c].h, false, bf.h,
                                                          (short)0, acc3, false, false);
          }
        }
        // tanh chain rule, elementwise on matching C/D layouts; store f16
        #pragma unroll
        for (int r = 0; r < 8; ++r) {
          const int p = r + 8 * lh;
          float a = fast_tanh(accv[r]);
          float s1 = 1.0f - a * a;
          act[(0 * 16 + p) * 128 + col] = (_Float16)a;
          if constexpr (S >= 2) {
            float zx = acc1[r];
            float ax = s1 * zx;
            act[(1 * 16 + p) * 128 + col] = (_Float16)ax;
            if constexpr (S == 4) {
              act[(2 * 16 + p) * 128 + col] = (_Float16)(s1 * acc2[r]);
              act[(3 * 16 + p) * 128 + col] =
                  (_Float16)(s1 * acc3[r] - 2.0f * a * zx * ax);
            }
          }
        }
      }
    }

    // ---------------- output layer (128 -> 2, zero-padded to 16) ----------------
    #pragma unroll
    for (int s = 0; s < S; ++s) {
      if (S == 4 && s == 1) continue;  // d/dx output unused by the PDE residual
      Frag16 af[4];
      #pragma unroll
      for (int c = 0; c < 4; ++c) {
        const _Float16* ap = act + (s * 16 + m) * 128 + c * 32 + 8 * lh;
        af[c].u[0] = *(const uint4*)ap;
        af[c].u[1] = *(const uint4*)(ap + 16);
      }
      v8f acc{};
      if (s == 0 && m < 2) {
        float bb = b5lds[m];
        #pragma unroll
        for (int r = 0; r < 8; ++r) acc[r] = bb;
      }
      #pragma unroll
      for (int c = 0; c < 4; ++c) {
        Frag16 bf;
        const _Float16* bp = w5lds + m * 128 + c * 32 + 8 * lh;
        bf.u[0] = *(const uint4*)bp;
        bf.u[1] = *(const uint4*)(bp + 16);
        acc = __builtin_amdgcn_wmma_f32_16x16x32_f16(false, af[c].h, false, bf.h,
                                                     (short)0, acc, false, false);
      }
      if (m < 2) {  // only cols 0,1 are real outputs
        #pragma unroll
        for (int r = 0; r < 8; ++r) stage[s * 32 + (r + 8 * lh) * 2 + m] = acc[r];
      }
    }

    // ---------------- final assembly + global stores ----------------
    {
      const int p = lane >> 1, comp = lane & 1;  // 16 points x 2 components
      if constexpr (S == 4) {
        float hr = stage[0 * 32 + p * 2 + 0], hi = stage[0 * 32 + p * 2 + 1];
        float htr = stage[2 * 32 + p * 2 + 0], hti = stage[2 * 32 + p * 2 + 1];
        float hxxr = stage[3 * 32 + p * 2 + 0], hxxi = stage[3 * 32 + p * 2 + 1];
        float mag2 = hr * hr + hi * hi;
        out_h[(pbase + p) * 2 + comp] = comp ? hi : hr;
        float res = comp ? (htr + 0.5f * hxxi + mag2 * hi)
                         : (-hti + 0.5f * hxxr + mag2 * hr);
        out_aux[(pbase + p) * 2 + comp] = res;
      } else if constexpr (S == 2) {
        out_h[(pbase + p) * 2 + comp]   = stage[0 * 32 + p * 2 + comp];
        out_aux[(pbase + p) * 2 + comp] = stage[1 * 32 + p * 2 + comp];
      } else {
        out_h[(pbase + p) * 2 + comp] = stage[p * 2 + comp];
      }
    }
  }
}

// ---------------------------------------------------------------------------
extern "C" void kernel_launch(void* const* d_in, const int* in_sizes, int n_in,
                              void* d_out, int out_size, void* d_ws, size_t ws_size,
                              hipStream_t stream) {
  const float* tx_eq   = (const float*)d_in[0];
  const float* tx_init = (const float*)d_in[1];
  const float* tx_bnd  = (const float*)d_in[2];
  const float* w0 = (const float*)d_in[3];
  const float* b0 = (const float*)d_in[4];
  const float* w1 = (const float*)d_in[5];
  const float* b1 = (const float*)d_in[6];
  const float* w2 = (const float*)d_in[7];
  const float* b2 = (const float*)d_in[8];
  const float* w3 = (const float*)d_in[9];
  const float* b3 = (const float*)d_in[10];
  const float* w4 = (const float*)d_in[11];
  const float* b4 = (const float*)d_in[12];
  const float* w5 = (const float*)d_in[13];
  const float* b5 = (const float*)d_in[14];

  _Float16* wpack = (_Float16*)d_ws;
  float* out = (float*)d_out;
  float* h_eq   = out;                 // [200000,2]
  float* resid  = out + 2 * 200000;    // [200000,2]
  float* h_init = out + 4 * 200000;    // [32768,2]
  float* h_bnd  = h_init + 2 * 32768;  // [32768,2]
  float* dhbdx  = h_bnd + 2 * 32768;   // [32768,2]

  prep_weights_kernel<<<(4 * 128 * 128 + 16 * 128 + 255) / 256, 256, 0, stream>>>(
      w1, w2, w3, w4, w5, wpack);

  auto smem_bytes = [](int S) -> size_t {
    return (size_t)(4 * 128 * 128 + 16 * 128) * 2   // f16 weights
         + 900 * 4                                  // biases + w0/b0
         + (size_t)NW * S * 16 * 128 * 2            // f16 activation scratch
         + (size_t)NW * S * 16 * 2 * 4;             // f32 output staging
  };

  // 200000/16 = 12500 tiles
  const int eq_tiles = 12500, small_tiles = 2048;
  pinn_mlp_kernel<4><<<(eq_tiles + NW - 1) / NW, NW * WAVE, smem_bytes(4), stream>>>(
      tx_eq, wpack, w0, b0, b1, b2, b3, b4, b5, h_eq, resid, eq_tiles);
  pinn_mlp_kernel<1><<<small_tiles / NW, NW * WAVE, smem_bytes(1), stream>>>(
      tx_init, wpack, w0, b0, b1, b2, b3, b4, b5, h_init, nullptr, small_tiles);
  pinn_mlp_kernel<2><<<small_tiles / NW, NW * WAVE, smem_bytes(2), stream>>>(
      tx_bnd, wpack, w0, b0, b1, b2, b3, b4, b5, h_bnd, dhbdx, small_tiles);
}
